// SpikeTransformerBlock_53463752901305
// MI455X (gfx1250) — matrix operand order repaired and checked by reading, not compiled, not run on hardware
//
#include <hip/hip_runtime.h>

// ---------------- problem constants (fixed by reference; cur_pos == S) ----------------
#define T_ 4
#define B_ 8
#define S_ 1024
#define D_ 768
#define HF_ 3072
#define H_ 12
#define HD_ 64

#define SD_   (S_ * D_)           // 786432
#define SH_   (S_ * HF_)          // 3145728
#define BSD_  (B_ * SD_)          // 6291456   (per-timestep stride, D-shaped)
#define BSH_  (B_ * SH_)          // 25165824  (per-timestep stride, Hf-shaped)
#define E_    (T_ * BSD_)         // 25165824
#define EH_   (T_ * BSH_)         // 100663296
#define M_    (T_ * B_ * S_)      // 32768 GEMM rows

// stat slots: 0=Q 1=V 2=K 3=ATT 4=O1 5=O2 ; stats[slot*16 + b]=sum, [slot*16+8+b]=sumsq
#define ST_Q   0
#define ST_V   1
#define ST_K   2
#define ST_ATT 3
#define ST_O1  4
#define ST_O2  5

typedef __bf16 bf16x8  __attribute__((ext_vector_type(8)));
typedef __bf16 bf16x16 __attribute__((ext_vector_type(16)));
typedef float  f32x8   __attribute__((ext_vector_type(8)));
typedef int    i32x4   __attribute__((ext_vector_type(4)));

#define AS1 __attribute__((address_space(1)))
#define AS3 __attribute__((address_space(3)))

// ---- async global->LDS copy (ASYNCcnt) with compile-time fallback to load + ds_store ----
#if __has_builtin(__builtin_amdgcn_global_load_async_to_lds_b128)
#define HAVE_ASYNC 1
#define CPY16(g, l) __builtin_amdgcn_global_load_async_to_lds_b128( \
    (AS1 i32x4*)(AS1 void*)(g), (AS3 i32x4*)(AS3 void*)(l), 0, 0)
#else
#define HAVE_ASYNC 0
#define CPY16(g, l) (*(bf16x8*)(l) = *(const bf16x8*)(g))
#endif

#if HAVE_ASYNC
#if __has_builtin(__builtin_amdgcn_s_wait_asynccnt)
#define WAIT_ASYNC(n) __builtin_amdgcn_s_wait_asynccnt(n)
#else
#define WAIT_ASYNC(n) asm volatile("s_wait_asynccnt %0" ::"i"(n) : "memory")
#endif
#else
#define WAIT_ASYNC(n) ((void)0)
#endif

__device__ __forceinline__ __bf16 f2bf(float f) {
    unsigned u = __float_as_uint(f);
    unsigned r = (u + 0x7fffu + ((u >> 16) & 1u)) >> 16;   // round-to-nearest-even
    unsigned short h = (unsigned short)r;
    __bf16 b;
    __builtin_memcpy(&b, &h, 2);
    return b;
}
__device__ __forceinline__ float bf2f(__bf16 b) {
    unsigned short h;
    __builtin_memcpy(&h, &b, 2);
    return __uint_as_float(((unsigned)h) << 16);
}

// ---------------- WMMA fragments (wave32, v_wmma_f32_16x16x32_bf16) ----------------
union Frag {
    bf16x16 v;
    struct { bf16x8 lo; bf16x8 hi; } p;
};

// LDS tile row stride: 40 bf16 (80B = 20 dwords); 20*r mod 64 distinct for r=0..15 -> no bank conflicts
#define LPAD 40
#define LTILE (128 * LPAD)

// A-frag from LDS tile: lane = row M=lane&15; K chunks [kb..kb+7],[kb+16..kb+23], kb=(lane>>4)*8
__device__ __forceinline__ bf16x16 lds_afrag(const __bf16* s, int row0, int lane) {
    Frag f;
    const __bf16* p = s + (row0 + (lane & 15)) * LPAD + ((lane >> 4) << 3);
    f.p.lo = *(const bf16x8*)p;
    f.p.hi = *(const bf16x8*)(p + 16);
    return f.v;
}
// B-frag from LDS tile of transposed weights (N x K): lane = col N=lane&15;
// K chunks [kb..kb+7],[kb+16..kb+23], kb=(lane>>4)*8
__device__ __forceinline__ bf16x16 lds_bfrag(const __bf16* s, int n0, int lane) {
    Frag f;
    const __bf16* p = s + (n0 + (lane & 15)) * LPAD + ((lane >> 4) << 3);
    f.p.lo = *(const bf16x8*)p;
    f.p.hi = *(const bf16x8*)(p + 16);
    return f.v;
}

// stage one K=32 slab of A (128x32) + Bt hi/lo (128x32 each) into LDS; 6 x 16B per thread
__device__ __forceinline__ void stage_load(const __bf16* __restrict__ A,
                                           const __bf16* __restrict__ BtH,
                                           const __bf16* __restrict__ BtL,
                                           __bf16* sA, __bf16* sBh, __bf16* sBl,
                                           int K, int rowBaseA, int rowBaseB,
                                           int kk, int tid) {
#pragma unroll
    for (int c = tid; c < 512; c += 256) {
        int row = c >> 2, c4 = (c & 3) << 3;        // 4 x 8-elem chunks per row
        int lo = row * LPAD + c4;
        size_t go = (size_t)row * K + kk + c4;
        CPY16(A   + (size_t)rowBaseA * K + go, sA  + lo);
        CPY16(BtH + (size_t)rowBaseB * K + go, sBh + lo);
        CPY16(BtL + (size_t)rowBaseB * K + go, sBl + lo);
    }
}

// ---------------- GEMM: C[M,N] = A[M,K] @ (BtHi+BtLo)^T (+bias) ----------------
// A bf16 row-major (binary spikes -> exact). Weights pre-split hi/lo bf16, transposed (N x K):
// two bf16 WMMAs into one f32 accumulator recover ~fp32 accuracy.
// Block 256 thr = 8 waves (2 x 4); block tile 128x128; wave tile 64x32; K staged in LDS,
// double-buffered, filled via async-to-LDS and overlapped with WMMA via s_wait_asynccnt.
__global__ __launch_bounds__(256)
void k_gemm(const __bf16* __restrict__ A, const __bf16* __restrict__ BtH,
            const __bf16* __restrict__ BtL, float* __restrict__ C,
            const float* __restrict__ bias, int N, int K) {
    __shared__ __bf16 sA[2][LTILE];
    __shared__ __bf16 sBh[2][LTILE];
    __shared__ __bf16 sBl[2][LTILE];

    const int tid  = threadIdx.x;
    const int lane = tid & 31;
    const int wid  = tid >> 5;
    const int wmRow = (wid >> 2) * 64;   // wave row base within block tile
    const int wnCol = (wid & 3) * 32;    // wave col base within block tile
    const int rowBaseA = blockIdx.y * 128;
    const int rowBaseB = blockIdx.x * 128;

    f32x8 acc[4][2];
#pragma unroll
    for (int i = 0; i < 4; ++i)
#pragma unroll
        for (int j = 0; j < 2; ++j)
            acc[i][j] = f32x8{0.f, 0.f, 0.f, 0.f, 0.f, 0.f, 0.f, 0.f};

    const int nStages = K >> 5;
    stage_load(A, BtH, BtL, sA[0], sBh[0], sBl[0], K, rowBaseA, rowBaseB, 0, tid);

    for (int s = 0; s < nStages; ++s) {
        if (s + 1 < nStages) {
            stage_load(A, BtH, BtL, sA[(s + 1) & 1], sBh[(s + 1) & 1], sBl[(s + 1) & 1],
                       K, rowBaseA, rowBaseB, (s + 1) << 5, tid);
            WAIT_ASYNC(6);   // 6 per stage, in-order => stage s landed
        } else {
            WAIT_ASYNC(0);
        }
        __syncthreads();

        const __bf16* a_s  = sA[s & 1];
        const __bf16* bh_s = sBh[s & 1];
        const __bf16* bl_s = sBl[s & 1];
        bf16x16 a[4], bh[2], bl[2];
#pragma unroll
        for (int i = 0; i < 4; ++i) a[i] = lds_afrag(a_s, wmRow + 16 * i, lane);
#pragma unroll
        for (int j = 0; j < 2; ++j) {
            bh[j] = lds_bfrag(bh_s, wnCol + 16 * j, lane);
            bl[j] = lds_bfrag(bl_s, wnCol + 16 * j, lane);
        }
#pragma unroll
        for (int i = 0; i < 4; ++i)
#pragma unroll
            for (int j = 0; j < 2; ++j) {
                acc[i][j] = __builtin_amdgcn_wmma_f32_16x16x32_bf16(
                    false, a[i], false, bh[j], (short)0, acc[i][j], false, false);
                acc[i][j] = __builtin_amdgcn_wmma_f32_16x16x32_bf16(
                    false, a[i], false, bl[j], (short)0, acc[i][j], false, false);
            }
        __syncthreads();   // stage s fully consumed before its buffer is refilled
    }

    const int bm = rowBaseA + wmRow;
    const int bn = rowBaseB + wnCol;
    const int colIn = lane & 15;
    const int rowHalf = (lane >> 4) * 8;
#pragma unroll
    for (int i = 0; i < 4; ++i)
#pragma unroll
        for (int j = 0; j < 2; ++j) {
            int col = bn + 16 * j + colIn;
            float bv = bias ? bias[col] : 0.f;
#pragma unroll
            for (int e = 0; e < 8; ++e) {
                int row = bm + 16 * i + rowHalf + e;
                C[(size_t)row * N + col] = acc[i][j][e] + bv;
            }
        }
}

// ---------------- elementwise / BN / IF kernels ----------------
__global__ __launch_bounds__(256) void k_zero_stats(float* stats) {
    if (threadIdx.x < 128) stats[threadIdx.x] = 0.f;
}

// IF scan over T on x -> binary spikes (bf16)
__global__ __launch_bounds__(256) void k_if_x(const float* __restrict__ x,
                                              __bf16* __restrict__ sp) {
    size_t i = (size_t)blockIdx.x * 256 + threadIdx.x;  // over B*S*D
    float v = 0.f;
#pragma unroll
    for (int t = 0; t < T_; ++t) {
        size_t idx = i + (size_t)t * BSD_;
        v += x[idx];
        float s = (v >= 1.f) ? 1.f : 0.f;
        sp[idx] = f2bf(s);
        v *= (1.f - s);
    }
}

// transpose + hi/lo bf16 split of a (K x N) fp32 weight -> Wt (N x K)
__global__ __launch_bounds__(256) void k_wt(const float* __restrict__ W,
                                            __bf16* __restrict__ Hi, __bf16* __restrict__ Lo,
                                            int K, int N) {
    size_t i = (size_t)blockIdx.x * 256 + threadIdx.x;  // over N*K
    if (i >= (size_t)K * N) return;
    int n = (int)(i / K), k = (int)(i % K);
    float w  = W[(size_t)k * N + n];
    __bf16 h = f2bf(w);
    Hi[i] = h;
    Lo[i] = f2bf(w - bf2f(h));
}

// per-b sum/sumsq over (T,S,Dd) with f32 atomics
__global__ __launch_bounds__(256) void k_stats(const float* __restrict__ X, float* stats,
                                               int slot, int Dd) {
    int b = blockIdx.y;
    size_t per_t = (size_t)S_ * Dd;
    size_t NE = (size_t)T_ * per_t;
    float s = 0.f, s2 = 0.f;
    for (size_t i = (size_t)blockIdx.x * 256 + threadIdx.x; i < NE; i += (size_t)gridDim.x * 256) {
        size_t t = i / per_t, r = i % per_t;
        float v = X[(t * B_ + b) * per_t + r];
        s += v; s2 += v * v;
    }
    __shared__ float sh[256], sh2[256];
    sh[threadIdx.x] = s; sh2[threadIdx.x] = s2;
    __syncthreads();
    for (int off = 128; off > 0; off >>= 1) {
        if ((int)threadIdx.x < off) { sh[threadIdx.x] += sh[threadIdx.x + off];
                                      sh2[threadIdx.x] += sh2[threadIdx.x + off]; }
        __syncthreads();
    }
    if (threadIdx.x == 0) {
        atomicAdd(&stats[slot * 16 + b], sh[0]);
        atomicAdd(&stats[slot * 16 + 8 + b], sh2[0]);
    }
}

__device__ __forceinline__ void load_stat(const float* stats, int slot, int b, float invN,
                                          float& m, float& rs) {
    float s = stats[slot * 16 + b], s2 = stats[slot * 16 + 8 + b];
    m = s * invN;
    float var = s2 * invN - m * m;
    rs = rsqrtf(var + 1e-5f);
}

// BN + IF for Q, V, K with membrane potential carried Q -> V -> K (reference semantics)
__global__ __launch_bounds__(256)
void k_bn_if_qvk(const float* __restrict__ Qp, const float* __restrict__ Vp,
                 const float* __restrict__ Kp, const float* __restrict__ stats,
                 __bf16* __restrict__ Qs, __bf16* __restrict__ Vs, __bf16* __restrict__ Ks) {
    size_t i = (size_t)blockIdx.x * 256 + threadIdx.x;  // over B*S*D
    int b = (int)(i / SD_);
    const float invN = 1.f / (float)((size_t)T_ * SD_);
    float mq, rq, mv, rv, mk, rk;
    load_stat(stats, ST_Q, b, invN, mq, rq);
    load_stat(stats, ST_V, b, invN, mv, rv);
    load_stat(stats, ST_K, b, invN, mk, rk);
    const float* src[3] = {Qp, Vp, Kp};
    __bf16* dst[3] = {Qs, Vs, Ks};
    float mean[3] = {mq, mv, mk}, rstd[3] = {rq, rv, rk};
    float v = 0.f;
#pragma unroll
    for (int w = 0; w < 3; ++w) {
#pragma unroll
        for (int t = 0; t < T_; ++t) {
            size_t idx = i + (size_t)t * BSD_;
            v += (src[w][idx] - mean[w]) * rstd[w];
            float s = (v >= 1.f) ? 1.f : 0.f;
            dst[w][idx] = f2bf(s);
            v *= (1.f - s);
        }
    }
}

// QK[t,b,h,hd] = sum_s Qs*Ks, then IF over t -> QKs (stored as (T,B,D) float)
__global__ __launch_bounds__(256)
void k_qk_if(const __bf16* __restrict__ Qs, const __bf16* __restrict__ Ks,
             float* __restrict__ QKs) {
    int b = blockIdx.x / H_, h = blockIdx.x % H_;
    int hd = threadIdx.x & 63, t = threadIdx.x >> 6;
    size_t base = ((size_t)t * B_ + b) * SD_ + h * HD_ + hd;
    float acc = 0.f;
    for (int s = 0; s < S_; ++s) {
        size_t idx = base + (size_t)s * D_;
        acc += bf2f(Qs[idx]) * bf2f(Ks[idx]);
    }
    __shared__ float sh[T_][HD_];
    sh[t][hd] = acc;
    __syncthreads();
    if (t == 0) {
        float v = 0.f;
#pragma unroll
        for (int tt = 0; tt < T_; ++tt) {
            v += sh[tt][hd];
            float s = (v >= 1.f) ? 1.f : 0.f;
            v *= (1.f - s);
            QKs[((size_t)tt * B_ + b) * D_ + h * HD_ + hd] = s;
        }
    }
}

// QKV[t,b,s,d] = Vs[t,b,s,d] * QKs[t,b,d]  (binary mask product, bf16 out)
__global__ __launch_bounds__(256)
void k_mask(const __bf16* __restrict__ Vs, const float* __restrict__ QKs,
            __bf16* __restrict__ QKV) {
    size_t i = (size_t)blockIdx.x * 256 + threadIdx.x;  // over E_
    size_t t = i / BSD_, r = i % BSD_;
    int b = (int)(r / SD_), d = (int)(i % D_);
    QKV[i] = f2bf(bf2f(Vs[i]) * QKs[(t * B_ + b) * D_ + d]);
}

// h = sp + bn(attp); write h (f32) and sp2 = IF(h) over t (bf16)
__global__ __launch_bounds__(256)
void k_bn_add_if(const float* __restrict__ attp, const __bf16* __restrict__ sp,
                 const float* __restrict__ stats, float* __restrict__ h,
                 __bf16* __restrict__ sp2) {
    size_t i = (size_t)blockIdx.x * 256 + threadIdx.x;  // over B*S*D
    int b = (int)(i / SD_);
    float m, rs;
    load_stat(stats, ST_ATT, b, 1.f / (float)((size_t)T_ * SD_), m, rs);
    float v = 0.f;
#pragma unroll
    for (int t = 0; t < T_; ++t) {
        size_t idx = i + (size_t)t * BSD_;
        float hv = bf2f(sp[idx]) + (attp[idx] - m) * rs;
        h[idx] = hv;
        v += hv;
        float s = (v >= 1.f) ? 1.f : 0.f;
        sp2[idx] = f2bf(s);
        v *= (1.f - s);
    }
}

// s1 = IF(bn(o1p)) over t (Hf-shaped)
__global__ __launch_bounds__(256)
void k_bn_if_o1(const float* __restrict__ o1p, const float* __restrict__ stats,
                __bf16* __restrict__ s1) {
    size_t i = (size_t)blockIdx.x * 256 + threadIdx.x;  // over B*S*HF
    int b = (int)(i / SH_);
    float m, rs;
    load_stat(stats, ST_O1, b, 1.f / (float)((size_t)T_ * SH_), m, rs);
    float v = 0.f;
#pragma unroll
    for (int t = 0; t < T_; ++t) {
        size_t idx = i + (size_t)t * BSH_;
        v += (o1p[idx] - m) * rs;
        float s = (v >= 1.f) ? 1.f : 0.f;
        s1[idx] = f2bf(s);
        v *= (1.f - s);
    }
}

// out = h + bn(o2p)
__global__ __launch_bounds__(256)
void k_final(const float* __restrict__ h, const float* __restrict__ o2p,
             const float* __restrict__ stats, float* __restrict__ out) {
    size_t i = (size_t)blockIdx.x * 256 + threadIdx.x;  // over B*S*D
    int b = (int)(i / SD_);
    float m, rs;
    load_stat(stats, ST_O2, b, 1.f / (float)((size_t)T_ * SD_), m, rs);
#pragma unroll
    for (int t = 0; t < T_; ++t) {
        size_t idx = i + (size_t)t * BSD_;
        out[idx] = h[idx] + (o2p[idx] - m) * rs;
    }
}

// ---------------- host launcher ----------------
extern "C" void kernel_launch(void* const* d_in, const int* in_sizes, int n_in,
                              void* d_out, int out_size, void* d_ws, size_t ws_size,
                              hipStream_t stream) {
    const float* x  = (const float*)d_in[0];
    const float* wq = (const float*)d_in[1];
    const float* wk = (const float*)d_in[2];
    const float* wv = (const float*)d_in[3];
    const float* wo = (const float*)d_in[4];
    const float* w1 = (const float*)d_in[5];
    const float* b1 = (const float*)d_in[6];
    const float* w2 = (const float*)d_in[7];
    const float* b2 = (const float*)d_in[8];
    float* out = (float*)d_out;

    char* ws = (char*)d_ws;
    size_t off = 0;
    auto alloc = [&](size_t bytes) -> void* {
        void* p = ws + off;
        off = (off + bytes + 255) & ~(size_t)255;
        return p;
    };
    const size_t W768 = (size_t)D_ * D_;      // 589824
    const size_t W1N  = (size_t)D_ * HF_;     // 2359296

    __bf16* wqH = (__bf16*)alloc(W768 * 2); __bf16* wqL = (__bf16*)alloc(W768 * 2);
    __bf16* wkH = (__bf16*)alloc(W768 * 2); __bf16* wkL = (__bf16*)alloc(W768 * 2);
    __bf16* wvH = (__bf16*)alloc(W768 * 2); __bf16* wvL = (__bf16*)alloc(W768 * 2);
    __bf16* woH = (__bf16*)alloc(W768 * 2); __bf16* woL = (__bf16*)alloc(W768 * 2);
    __bf16* w1H = (__bf16*)alloc(W1N * 2);  __bf16* w1L = (__bf16*)alloc(W1N * 2);
    __bf16* w2H = (__bf16*)alloc(W1N * 2);  __bf16* w2L = (__bf16*)alloc(W1N * 2);
    __bf16* sp  = (__bf16*)alloc((size_t)E_ * 2);
    float*  Qp  = (float*)alloc((size_t)E_ * 4);
    float*  Kp  = (float*)alloc((size_t)E_ * 4);
    float*  Vp  = (float*)alloc((size_t)E_ * 4);
    __bf16* Qs  = (__bf16*)alloc((size_t)E_ * 2);
    __bf16* Ks  = (__bf16*)alloc((size_t)E_ * 2);
    __bf16* Vs  = (__bf16*)alloc((size_t)E_ * 2);
    float*  stats = (float*)alloc(1024);
    float*  QKs = (float*)alloc((size_t)T_ * B_ * D_ * 4);
    float*  o1p = (float*)alloc((size_t)EH_ * 4);
    __bf16* s1  = (__bf16*)alloc((size_t)EH_ * 2);
    // aliases (lifetimes are disjoint)
    __bf16* QKV  = (__bf16*)Qp;   // Qp preact dead after k_bn_if_qvk
    float*  attp = Kp;            // Kp preact dead after k_bn_if_qvk
    float*  hbuf = Vp;            // Vp preact dead after k_bn_if_qvk
    __bf16* sp2  = Qs;            // Qs dead after k_qk_if
    float*  o2p  = o1p;           // o1p dead after k_bn_if_o1

    const dim3 blk(256);
    auto gemm = [&](const __bf16* A, const __bf16* BH, const __bf16* BL, float* C,
                    const float* bias, int N, int K) {
        dim3 g(N / 128, M_ / 128);
        k_gemm<<<g, blk, 0, stream>>>(A, BH, BL, C, bias, N, K);
    };

    k_zero_stats<<<1, blk, 0, stream>>>(stats);
    k_if_x<<<BSD_ / 256, blk, 0, stream>>>(x, sp);

    k_wt<<<(unsigned)(W768 / 256), blk, 0, stream>>>(wq, wqH, wqL, D_, D_);
    k_wt<<<(unsigned)(W768 / 256), blk, 0, stream>>>(wk, wkH, wkL, D_, D_);
    k_wt<<<(unsigned)(W768 / 256), blk, 0, stream>>>(wv, wvH, wvL, D_, D_);
    k_wt<<<(unsigned)(W768 / 256), blk, 0, stream>>>(wo, woH, woL, D_, D_);
    k_wt<<<(unsigned)(W1N / 256), blk, 0, stream>>>(w1, w1H, w1L, D_, HF_);
    k_wt<<<(unsigned)(W1N / 256), blk, 0, stream>>>(w2, w2H, w2L, HF_, D_);

    gemm(sp, wqH, wqL, Qp, nullptr, D_, D_);
    gemm(sp, wvH, wvL, Vp, nullptr, D_, D_);
    gemm(sp, wkH, wkL, Kp, nullptr, D_, D_);

    dim3 gs(1024, B_);
    k_stats<<<gs, blk, 0, stream>>>(Qp, stats, ST_Q, D_);
    k_stats<<<gs, blk, 0, stream>>>(Vp, stats, ST_V, D_);
    k_stats<<<gs, blk, 0, stream>>>(Kp, stats, ST_K, D_);

    k_bn_if_qvk<<<BSD_ / 256, blk, 0, stream>>>(Qp, Vp, Kp, stats, Qs, Vs, Ks);
    k_qk_if<<<B_ * H_, blk, 0, stream>>>(Qs, Ks, QKs);
    k_mask<<<E_ / 256, blk, 0, stream>>>(Vs, QKs, QKV);

    gemm(QKV, woH, woL, attp, nullptr, D_, D_);
    k_stats<<<gs, blk, 0, stream>>>(attp, stats, ST_ATT, D_);
    k_bn_add_if<<<BSD_ / 256, blk, 0, stream>>>(attp, sp, stats, hbuf, sp2);

    gemm(sp2, w1H, w1L, o1p, b1, HF_, D_);
    k_stats<<<gs, blk, 0, stream>>>(o1p, stats, ST_O1, HF_);
    k_bn_if_o1<<<BSH_ / 256, blk, 0, stream>>>(o1p, stats, s1);

    gemm(s1, w2H, w2L, o2p, b2, D_, HF_);
    k_stats<<<gs, blk, 0, stream>>>(o2p, stats, ST_O2, D_);
    k_final<<<BSD_ / 256, blk, 0, stream>>>(hbuf, o2p, stats, out);
}
